// Decoder_54305566491078
// MI455X (gfx1250) — matrix operand order repaired and checked by reading, not compiled, not run on hardware
//
#include <hip/hip_runtime.h>
#include <math.h>

// ---------------------------------------------------------------------------
// Types for CDNA5 WMMA
// ---------------------------------------------------------------------------
typedef __attribute__((ext_vector_type(16))) __bf16        v16bf;
typedef __attribute__((ext_vector_type(8)))  float         v8f;
typedef __attribute__((ext_vector_type(4)))  unsigned int  u32x4;

union FragU {
    v16bf v;
    u32x4 q[2];
};

__device__ __forceinline__ unsigned short f2bf(float x) {
    unsigned u = __float_as_uint(x);
    unsigned r = 0x7FFFu + ((u >> 16) & 1u);
    u += r;
    return (unsigned short)(u >> 16);
}

__device__ __forceinline__ float sigf(float x) {
    return 1.0f / (1.0f + expf(-x));
}

// ---------------------------------------------------------------------------
// Fragment loads: A = 2 contiguous b128 per lane, each B tile = 2 b128.
// ---------------------------------------------------------------------------
template<int NT>
__device__ __forceinline__ void load_frags(const unsigned short* __restrict__ pa,
                                           const unsigned short* __restrict__ pb,
                                           FragU& A, FragU (&B)[NT])
{
    A.q[0] = *(const u32x4*)(pa);
    A.q[1] = *(const u32x4*)(pa + 8);
#pragma unroll
    for (int j = 0; j < NT; ++j) {
        const unsigned short* p = pb + (size_t)j * 512;
        B[j].q[0] = *(const u32x4*)(p);
        B[j].q[1] = *(const u32x4*)(p + 8);
    }
}

template<int NT>
__device__ __forceinline__ void mma_nt(const FragU& A, const FragU (&B)[NT],
                                       v8f (&acc)[NT])
{
#pragma unroll
    for (int j = 0; j < NT; ++j)     // NT independent WMMAs, no RAW between them
        acc[j] = __builtin_amdgcn_wmma_f32_16x16x32_bf16(
                     false, A.v, false, B[j].v, (short)0, acc[j], false, false);
}

// Branchless K-loop over one A segment, 2x ping-pong software pipeline
// (no register rotation copies -> no v_mov/v_nop hazard filler).
// pa: per-lane A pointer at segment start (advances 32 elems/chunk).
// pb: per-lane B pointer (advances bstride/chunk); returns pb after segment.
template<int NT>
__device__ __forceinline__ const unsigned short*
seg_mm(const unsigned short* __restrict__ pa, int chunks,
       const unsigned short* pb, size_t bstride, v8f (&acc)[NT])
{
    const unsigned short* pb_ret = pb + (size_t)chunks * bstride;
    FragU A0, A1;
    FragU B0[NT], B1[NT];

    if (chunks & 1) {                       // peel odd chunk (prenet1 only)
        load_frags<NT>(pa, pb, A0, B0);
        mma_nt<NT>(A0, B0, acc);
        pa += 32; pb += bstride;
        chunks &= ~1;
        if (!chunks) return pb_ret;
    }

    load_frags<NT>(pa, pb, A0, B0);         // chunk 0 of the even part
    for (int c = 0; c < chunks - 2; c += 2) {
        load_frags<NT>(pa + 32, pb + bstride, A1, B1);       // load c+1
        __builtin_prefetch(pb + 2 * bstride, 0, 1);          // prefetch c+2
        mma_nt<NT>(A0, B0, acc);                             // compute c
        load_frags<NT>(pa + 64, pb + 2 * bstride, A0, B0);   // load c+2
        mma_nt<NT>(A1, B1, acc);                             // compute c+1
        pa += 64; pb += 2 * bstride;
    }
    load_frags<NT>(pa + 32, pb + bstride, A1, B1);           // last chunk
    mma_nt<NT>(A0, B0, acc);
    mma_nt<NT>(A1, B1, acc);
    return pb_ret;
}

// ---------------------------------------------------------------------------
// Generic skinny GEMM:  C[M,Npad] = act( A[M,K]_bf16 @ Bpacked[K,Npad]_bf16 + bias )
// A given as up to 3 row-major bf16 segments along K (widths multiple of 32).
// B pre-packed in per-lane 32x16 WMMA fragment layout (chunk-major).
// Each wave owns one 16-row M tile and NT adjacent 16-col N tiles.
// ACT: 0 = none, 1 = relu, 2 = tanh
// ---------------------------------------------------------------------------
template<int ACT, int NT>
__global__ __launch_bounds__(128)
void gemm_bf16_wmma(const unsigned short* __restrict__ a0, int k0,
                    const unsigned short* __restrict__ a1, int k1,
                    const unsigned short* __restrict__ a2, int k2,
                    const unsigned short* __restrict__ Bp,
                    int ntiles, int mtiles,
                    const float* __restrict__ bias,
                    float* __restrict__ Cf,
                    unsigned short* __restrict__ Cb,
                    int ldc)
{
    int ngroups = ntiles / NT;
    int wave    = blockIdx.x * (blockDim.x >> 5) + (threadIdx.x >> 5);
    int total   = mtiles * ngroups;
    if (wave >= total) return;            // wave-uniform: EXEC stays all-ones
    int mt   = wave / ngroups;
    int ng   = wave - mt * ngroups;
    int lane = threadIdx.x & 31;

    int arow  = mt * 16 + (lane & 15);
    int akoff = (lane >> 4) << 4;         // lanes 16..31 hold K+16..K+31

    v8f zero = {};
    v8f acc[NT];
#pragma unroll
    for (int j = 0; j < NT; ++j) acc[j] = zero;

    size_t bstride = (size_t)ntiles * 512;
    const unsigned short* pb = Bp + (size_t)(ng * NT) * 512 + (size_t)lane * 16;

    int c0 = k0 >> 5, c1 = k1 >> 5, c2 = k2 >> 5;
    if (c0) pb = seg_mm<NT>(a0 + (size_t)arow * k0 + akoff, c0, pb, bstride, acc);
    if (c1) pb = seg_mm<NT>(a1 + (size_t)arow * k1 + akoff, c1, pb, bstride, acc);
    if (c2) pb = seg_mm<NT>(a2 + (size_t)arow * k2 + akoff, c2, pb, bstride, acc);

    int rbase = mt * 16 + ((lane >> 4) << 3);
#pragma unroll
    for (int j = 0; j < NT; ++j) {
        int col  = (ng * NT + j) * 16 + (lane & 15);
        float bv = bias ? bias[col] : 0.0f;
#pragma unroll
        for (int r = 0; r < 8; ++r) {
            float v = acc[j][r] + bv;
            if (ACT == 1) v = fmaxf(v, 0.0f);
            if (ACT == 2) v = tanhf(v);
            if (Cf) Cf[(size_t)(rbase + r) * ldc + col] = v;
            if (Cb) Cb[(size_t)(rbase + r) * ldc + col] = f2bf(v);
        }
    }
}

// ---------------------------------------------------------------------------
// Weight packing: fp32 [Ksrc,Nsrc] (two stacked sources along K) -> bf16
// per-lane WMMA B-fragment layout, zero padded to [Kpad,Npad].
// ---------------------------------------------------------------------------
__global__ void pack_b_kernel(const float* __restrict__ s0, int k0rows,
                              const float* __restrict__ s1,
                              int Ksrc, int Nsrc, int Kpad, int Npad,
                              unsigned short* __restrict__ dst)
{
    int idx = blockIdx.x * blockDim.x + threadIdx.x;
    int total = Kpad * Npad;
    if (idx >= total) return;
    int tile = idx >> 9;
    int rem  = idx & 511;
    int lane = rem >> 4;
    int i    = rem & 15;
    int ntiles = Npad >> 4;
    int chunk  = tile / ntiles;
    int nt     = tile - chunk * ntiles;
    int n = nt * 16 + (lane & 15);
    int k = chunk * 32 + ((lane >> 4) << 4) + i;
    float v = 0.0f;
    if (k < Ksrc && n < Nsrc)
        v = (k < k0rows) ? s0[(size_t)k * Nsrc + n]
                         : s1[(size_t)(k - k0rows) * Nsrc + n];
    dst[idx] = f2bf(v);
}

// frame_w [1536,80] and stop_w [1536,1] concatenated along N -> [1536,96]
__global__ void pack_proj_kernel(const float* __restrict__ frame_w,
                                 const float* __restrict__ stop_w,
                                 unsigned short* __restrict__ dst)
{
    int idx = blockIdx.x * blockDim.x + threadIdx.x;
    const int Kpad = 1536, Npad = 96;
    if (idx >= Kpad * Npad) return;
    int tile = idx >> 9;
    int rem  = idx & 511;
    int lane = rem >> 4;
    int i    = rem & 15;
    int ntiles = Npad >> 4;
    int chunk  = tile / ntiles;
    int nt     = tile - chunk * ntiles;
    int n = nt * 16 + (lane & 15);
    int k = chunk * 32 + ((lane >> 4) << 4) + i;
    float v = 0.0f;
    if (n < 80)       v = frame_w[(size_t)k * 80 + n];
    else if (n == 80) v = stop_w[k];
    dst[idx] = f2bf(v);
}

__global__ void build_proj_bias(const float* __restrict__ frame_b,
                                const float* __restrict__ stop_b,
                                float* __restrict__ dst)
{
    int j = blockIdx.x * blockDim.x + threadIdx.x;
    if (j >= 96) return;
    dst[j] = (j < 80) ? frame_b[j] : (j == 80 ? stop_b[0] : 0.0f);
}

// decoder_inputs [B=32, 80, T=400] -> bf16 dec_in rows [t*32+b, 96] (go frame
// at t=0; cols 80..95 zero pad; row t holds decoder_inputs[:, :, t-1]).
__global__ void conv_decin_kernel(const float* __restrict__ dec,
                                  unsigned short* __restrict__ dst)
{
    int idx = blockIdx.x * blockDim.x + threadIdx.x;
    if (idx >= 12800 * 96) return;
    int row = idx / 96;
    int j   = idx - row * 96;
    int t   = row >> 5;
    int b   = row & 31;
    float v = 0.0f;
    if (t > 0 && j < 80)
        v = dec[((size_t)b * 80 + j) * 400 + (t - 1)];
    dst[idx] = f2bf(v);
}

__global__ void conv_f32_to_bf16(const float* __restrict__ src,
                                 unsigned short* __restrict__ dst, int n)
{
    int idx = blockIdx.x * blockDim.x + threadIdx.x;
    if (idx < n) dst[idx] = f2bf(src[idx]);
}

// LSTM gate nonlinearity. z [32,4096] (keras order i,f,g,o), c [32,1024].
__global__ void lstm_gates_kernel(const float* __restrict__ z,
                                  float* __restrict__ c,
                                  unsigned short* __restrict__ hbf)
{
    int idx = blockIdx.x * blockDim.x + threadIdx.x;
    if (idx >= 32 * 1024) return;
    int b = idx >> 10;
    int j = idx & 1023;
    const float* zb = z + (size_t)b * 4096;
    float ig = sigf(zb[j]);
    float fg = sigf(zb[j + 1024]);
    float gg = tanhf(zb[j + 2048]);
    float og = sigf(zb[j + 3072]);
    float c2 = fg * c[idx] + ig * gg;
    c[idx] = c2;
    hbf[idx] = f2bf(og * tanhf(c2));
}

// Location features: conv1d(K=31,SAME) over [att_w, att_w_cum] -> 32ch,
// then tanh(dense 32->128).  One thread per (b,t).
__global__ void loc_kernel(const float* __restrict__ attw,
                           const float* __restrict__ attwc,
                           const float* __restrict__ loc_conv,   // [31,2,32]
                           const float* __restrict__ loc_dense,  // [32,128]
                           float* __restrict__ locf)             // [B*Tm,128]
{
    int idx = blockIdx.x * blockDim.x + threadIdx.x;
    if (idx >= 32 * 200) return;
    int b = idx / 200;
    int t = idx - b * 200;
    float conv[32];
#pragma unroll
    for (int ch = 0; ch < 32; ++ch) conv[ch] = 0.0f;
    for (int k = 0; k < 31; ++k) {
        int tp = t + k - 15;
        if (tp < 0 || tp >= 200) continue;
        float w0 = attw[b * 200 + tp];
        float w1 = attwc[b * 200 + tp];
        const float* wk = loc_conv + k * 64;
#pragma unroll
        for (int ch = 0; ch < 32; ++ch)
            conv[ch] += w0 * wk[ch] + w1 * wk[32 + ch];
    }
    float* out = locf + (size_t)idx * 128;
    for (int d = 0; d < 128; ++d) {
        float s = 0.0f;
#pragma unroll
        for (int ch = 0; ch < 32; ++ch)
            s += conv[ch] * loc_dense[ch * 128 + d];
        out[d] = tanhf(s);
    }
}

// Energies + softmax + context.  One block per batch element.
__global__ __launch_bounds__(256)
void attention_kernel(const float* __restrict__ q,      // [32,128] (tanh'd)
                      const float* __restrict__ locf,   // [32*200,128]
                      const float* __restrict__ pm,     // [32*200,128]
                      const float* __restrict__ vvec,   // [128]
                      const float* __restrict__ memory, // [32,200,512]
                      float* __restrict__ attw,
                      float* __restrict__ attwc,
                      unsigned short* __restrict__ ctxbf,
                      float* __restrict__ aligns,       // [B,T,Tm]
                      int tstep)
{
    __shared__ float sE[200];
    __shared__ float sR[256];
    int tid = threadIdx.x;
    int b   = blockIdx.x;

    float e = -1e30f;
    if (tid < 200) {
        const float* qb = q + b * 128;
        const float* lf = locf + ((size_t)b * 200 + tid) * 128;
        const float* pf = pm   + ((size_t)b * 200 + tid) * 128;
        float s = 0.0f;
        for (int d = 0; d < 128; ++d)
            s += vvec[d] * tanhf(qb[d] + lf[d] + pf[d]);
        sE[tid] = s;
        e = s;
    }
    sR[tid] = e;
    __syncthreads();
    for (int s = 128; s > 0; s >>= 1) {
        if (tid < s) sR[tid] = fmaxf(sR[tid], sR[tid + s]);
        __syncthreads();
    }
    float m = sR[0];
    __syncthreads();
    float ex = 0.0f;
    if (tid < 200) { ex = expf(sE[tid] - m); sE[tid] = ex; }
    sR[tid] = ex;
    __syncthreads();
    for (int s = 128; s > 0; s >>= 1) {
        if (tid < s) sR[tid] += sR[tid + s];
        __syncthreads();
    }
    float inv = 1.0f / sR[0];
    __syncthreads();
    if (tid < 200) {
        float w = sE[tid] * inv;
        sE[tid] = w;
        attw[b * 200 + tid]  = w;
        attwc[b * 200 + tid] += w;
        aligns[((size_t)b * 400 + tstep) * 200 + tid] = w;
    }
    __syncthreads();
    for (int d = tid; d < 512; d += 256) {
        float s = 0.0f;
        for (int t = 0; t < 200; ++t)
            s += sE[t] * memory[((size_t)(b * 200 + t)) * 512 + d];
        ctxbf[b * 512 + d] = f2bf(s);
    }
}

// proj [32,96] -> mel_outputs [B,80,T] column t, gate [B,T] (sigmoid).
__global__ void emit_kernel(const float* __restrict__ proj,
                            float* __restrict__ mels,
                            float* __restrict__ gates,
                            int t)
{
    int idx = blockIdx.x * blockDim.x + threadIdx.x;
    if (idx >= 32 * 81) return;
    int b = idx / 81;
    int j = idx - b * 81;
    float v = proj[b * 96 + j];
    if (j < 80) mels[((size_t)b * 80 + j) * 400 + t] = v;
    else        gates[b * 400 + t] = 1.0f / (1.0f + expf(-v));
}

// ---------------------------------------------------------------------------
// Host side
// ---------------------------------------------------------------------------
extern "C" void kernel_launch(void* const* d_in, const int* in_sizes, int n_in,
                              void* d_out, int out_size, void* d_ws, size_t ws_size,
                              hipStream_t stream)
{
    (void)in_sizes; (void)n_in; (void)out_size; (void)ws_size;

    const float* memory   = (const float*)d_in[0];
    const float* dec      = (const float*)d_in[1];
    const float* pw0      = (const float*)d_in[2];
    const float* pb0      = (const float*)d_in[3];
    const float* pw1      = (const float*)d_in[4];
    const float* pb1      = (const float*)d_in[5];
    const float* l1k      = (const float*)d_in[6];
    const float* l1r      = (const float*)d_in[7];
    const float* l1b      = (const float*)d_in[8];
    const float* l2k      = (const float*)d_in[9];
    const float* l2r      = (const float*)d_in[10];
    const float* l2b      = (const float*)d_in[11];
    const float* wq       = (const float*)d_in[12];
    const float* wm       = (const float*)d_in[13];
    const float* vvec     = (const float*)d_in[14];
    const float* locc     = (const float*)d_in[15];
    const float* locd     = (const float*)d_in[16];
    const float* framew   = (const float*)d_in[17];
    const float* frameb   = (const float*)d_in[18];
    const float* stopw    = (const float*)d_in[19];
    const float* stopb    = (const float*)d_in[20];

    float* out    = (float*)d_out;
    float* mels   = out;                       // [32,80,400]
    float* gates  = out + (size_t)32 * 80 * 400;
    float* aligns = gates + (size_t)32 * 400;  // [32,400,200]

    char* ws = (char*)d_ws;
    size_t off = 0;
    auto take = [&](size_t bytes) -> void* {
        off = (off + 255) & ~(size_t)255;
        void* p = ws + off;
        off += bytes;
        return p;
    };

    unsigned short* W1p  = (unsigned short*)take((size_t)1792 * 4096 * 2);
    unsigned short* W2p  = (unsigned short*)take((size_t)2560 * 4096 * 2);
    unsigned short* Wqp  = (unsigned short*)take((size_t)1024 * 128 * 2);
    unsigned short* Wmp  = (unsigned short*)take((size_t)512 * 128 * 2);
    unsigned short* Wp0p = (unsigned short*)take((size_t)96 * 256 * 2);
    unsigned short* Wp1p = (unsigned short*)take((size_t)256 * 256 * 2);
    unsigned short* Wprp = (unsigned short*)take((size_t)1536 * 96 * 2);
    unsigned short* decin= (unsigned short*)take((size_t)12800 * 96 * 2);
    unsigned short* x1   = (unsigned short*)take((size_t)12800 * 256 * 2);
    unsigned short* xall = (unsigned short*)take((size_t)12800 * 256 * 2);
    unsigned short* membf= (unsigned short*)take((size_t)6400 * 512 * 2);
    float* pm    = (float*)take((size_t)6400 * 128 * 4);
    float* locf  = (float*)take((size_t)6400 * 128 * 4);
    float* z1    = (float*)take((size_t)32 * 4096 * 4);
    float* z2    = (float*)take((size_t)32 * 4096 * 4);
    float* qb    = (float*)take((size_t)32 * 128 * 4);
    float* proj  = (float*)take((size_t)32 * 96 * 4);
    float* pbias = (float*)take(96 * 4);
    unsigned short* h1bf = (unsigned short*)take((size_t)32 * 1024 * 2);
    unsigned short* h2bf = (unsigned short*)take((size_t)32 * 1024 * 2);
    unsigned short* ctxbf= (unsigned short*)take((size_t)32 * 512 * 2);
    float* c1    = (float*)take((size_t)32 * 1024 * 4);
    float* c2    = (float*)take((size_t)32 * 1024 * 4);
    float* attw  = (float*)take((size_t)32 * 200 * 4);
    float* attwc = (float*)take((size_t)32 * 200 * 4);

    auto cdiv = [](int a, int b) { return (a + b - 1) / b; };

    // --- zero the recurrent state every call (deterministic) ---
    hipMemsetAsync(h1bf, 0, (size_t)32 * 1024 * 2, stream);
    hipMemsetAsync(h2bf, 0, (size_t)32 * 1024 * 2, stream);
    hipMemsetAsync(ctxbf,0, (size_t)32 * 512 * 2, stream);
    hipMemsetAsync(c1,   0, (size_t)32 * 1024 * 4, stream);
    hipMemsetAsync(c2,   0, (size_t)32 * 1024 * 4, stream);
    hipMemsetAsync(attw, 0, (size_t)32 * 200 * 4, stream);
    hipMemsetAsync(attwc,0, (size_t)32 * 200 * 4, stream);

    // --- pack weights to bf16 WMMA fragment layout (L2-resident afterwards) ---
    pack_b_kernel<<<cdiv(1792 * 4096, 256), 256, 0, stream>>>(l1k, 768, l1r, 1792, 4096, 1792, 4096, W1p);
    pack_b_kernel<<<cdiv(2560 * 4096, 256), 256, 0, stream>>>(l2k, 1536, l2r, 2560, 4096, 2560, 4096, W2p);
    pack_b_kernel<<<cdiv(1024 * 128, 256), 256, 0, stream>>>(wq, 1024, nullptr, 1024, 128, 1024, 128, Wqp);
    pack_b_kernel<<<cdiv(512 * 128, 256), 256, 0, stream>>>(wm, 512, nullptr, 512, 128, 512, 128, Wmp);
    pack_b_kernel<<<cdiv(96 * 256, 256), 256, 0, stream>>>(pw0, 80, nullptr, 80, 256, 96, 256, Wp0p);
    pack_b_kernel<<<cdiv(256 * 256, 256), 256, 0, stream>>>(pw1, 256, nullptr, 256, 256, 256, 256, Wp1p);
    pack_proj_kernel<<<cdiv(1536 * 96, 256), 256, 0, stream>>>(framew, stopw, Wprp);
    build_proj_bias<<<1, 96, 0, stream>>>(frameb, stopb, pbias);

    // --- inputs to bf16 ---
    conv_decin_kernel<<<cdiv(12800 * 96, 256), 256, 0, stream>>>(dec, decin);
    conv_f32_to_bf16<<<cdiv(6400 * 512, 256), 256, 0, stream>>>(memory, membf, 6400 * 512);

    // --- hoisted batch GEMMs (parallel over all timesteps), NT=4 ---
    // Prenet layer 1: [12800,96] x [96,256], relu -> bf16
    gemm_bf16_wmma<1, 4><<<cdiv(800 * (16 / 4), 4), 128, 0, stream>>>(
        decin, 96, nullptr, 0, nullptr, 0, Wp0p, 16, 800, pb0, nullptr, x1, 256);
    // Prenet layer 2: [12800,256] x [256,256], relu -> bf16
    gemm_bf16_wmma<1, 4><<<cdiv(800 * (16 / 4), 4), 128, 0, stream>>>(
        x1, 256, nullptr, 0, nullptr, 0, Wp1p, 16, 800, pb1, nullptr, xall, 256);
    // processed_memory = tanh(memory @ wm): [6400,512] x [512,128]
    gemm_bf16_wmma<2, 4><<<cdiv(400 * (8 / 4), 4), 128, 0, stream>>>(
        membf, 512, nullptr, 0, nullptr, 0, Wmp, 8, 400, nullptr, pm, nullptr, 128);

    // --- sequential decode (NT=2 keeps 256 waves in flight per LSTM GEMM) ---
    for (int t = 0; t < 400; ++t) {
        const unsigned short* xt = xall + (size_t)t * 32 * 256;

        // LSTM1: [x(256)|ctx(512)|h1(1024)] x W1[1792,4096] + b
        gemm_bf16_wmma<0, 2><<<cdiv(2 * (256 / 2), 4), 128, 0, stream>>>(
            xt, 256, ctxbf, 512, h1bf, 1024, W1p, 256, 2, l1b, z1, nullptr, 4096);
        lstm_gates_kernel<<<cdiv(32 * 1024, 256), 256, 0, stream>>>(z1, c1, h1bf);

        // location features from previous att_w / att_w_cum
        loc_kernel<<<cdiv(32 * 200, 256), 256, 0, stream>>>(attw, attwc, locc, locd, locf);

        // q = tanh(att_h @ wq): [32,1024] x [1024,128]
        gemm_bf16_wmma<2, 4><<<cdiv(2 * (8 / 4), 4), 128, 0, stream>>>(
            h1bf, 1024, nullptr, 0, nullptr, 0, Wqp, 8, 2, nullptr, qb, nullptr, 128);

        // energies + softmax + context (+ alignment output)
        attention_kernel<<<32, 256, 0, stream>>>(
            qb, locf, pm, vvec, memory, attw, attwc, ctxbf, aligns, t);

        // LSTM2: [h1(1024)|ctx(512)|h2(1024)] x W2[2560,4096] + b
        gemm_bf16_wmma<0, 2><<<cdiv(2 * (256 / 2), 4), 128, 0, stream>>>(
            h1bf, 1024, ctxbf, 512, h2bf, 1024, W2p, 256, 2, l2b, z2, nullptr, 4096);
        lstm_gates_kernel<<<cdiv(32 * 1024, 256), 256, 0, stream>>>(z2, c2, h2bf);

        // frame/stop projection: [h2(1024)|ctx(512)] x [1536,96] + bias
        gemm_bf16_wmma<0, 2><<<cdiv(2 * (6 / 2), 4), 128, 0, stream>>>(
            h2bf, 1024, ctxbf, 512, nullptr, 0, Wprp, 6, 2, pbias, proj, nullptr, 96);
        emit_kernel<<<cdiv(32 * 81, 256), 256, 0, stream>>>(proj, mels, gates, t);
    }
}
